// MHA_43095701848407
// MI455X (gfx1250) — matrix operand order repaired and checked by reading, not compiled, not run on hardware
//
#include <hip/hip_runtime.h>

// ---------------------------------------------------------------------------
// MI455X (gfx1250) causal MHA forward: bf16 WMMA everywhere, f32 accumulate.
//   k1: qkv = x @ W_qkv  (+RoPE on q,k)  -> bf16 Q/K/V in (B,H,T,hd)
//   k2: flash attention per (B,H,128q)   -> bf16 y in (B*T, C)
//   k3: out = y @ W_proj  (A tile staged via Tensor Data Mover) -> f32 d_out
// ---------------------------------------------------------------------------

#define BATCH 4
#define SEQ   2048
#define CH    2048
#define NH    16
#define HD    128
#define NQKV  6144

typedef __attribute__((ext_vector_type(16))) __bf16 v16bf;
typedef __attribute__((ext_vector_type(8)))  __bf16 v8bf;
typedef __attribute__((ext_vector_type(8)))  float  v8f;

#if defined(__has_builtin)
#if __has_builtin(__builtin_amdgcn_tensor_load_to_lds)
#define HAVE_TDM 1
#else
#define HAVE_TDM 0
#endif
#else
#define HAVE_TDM 0
#endif

__device__ __forceinline__ __bf16 f2bf(float f) { return (__bf16)f; }

__device__ __forceinline__ v16bf cat8(v8bf lo, v8bf hi) {
  return __builtin_shufflevector(lo, hi, 0,1,2,3,4,5,6,7,8,9,10,11,12,13,14,15);
}

__device__ __forceinline__ v8f wmma_bf16(v16bf a, v16bf b, v8f c) {
  return __builtin_amdgcn_wmma_f32_16x16x32_bf16(false, a, false, b, (short)0, c,
                                                 false, false);
}

#if HAVE_TDM
// TDM: DMA a 128(row) x 32(col) bf16 tile out of an 8192x2048 bf16 tensor into
// LDS. pad_enable inserts 32 B after every 64 B row so the LDS image lands at
// pitch 48 elements (96 B) -- identical to the hand-staged layout.
__device__ __forceinline__ void tdm_load_tile_128x32(const __bf16* gsrc,
                                                     unsigned lds_off) {
  typedef __attribute__((ext_vector_type(4))) unsigned int v4u;
  typedef __attribute__((ext_vector_type(8))) int v8i;
  typedef __attribute__((ext_vector_type(4))) int v4i;
  const unsigned long long ga = (unsigned long long)(uintptr_t)gsrc;
  v4u g0;
  g0[0] = 1u;                                   // count=1 (valid user D#)
  g0[1] = lds_off;                              // lds_addr (bytes)
  g0[2] = (unsigned)(ga & 0xFFFFFFFFu);         // global_addr[31:0]
  g0[3] = (unsigned)(ga >> 32) | 0x80000000u;   // global_addr[56:32] | type=2
  v8i g1;
  // data_size=1 (2B) | pad_enable | pad_interval=3 (16 dw) | pad_amount=7 (8 dw)
  g1[0] = (int)((1u << 16) | (1u << 20) | (3u << 22) | (7u << 25));
  g1[1] = (int)(2048u << 16);                   // tensor_dim0 = 2048 (lo16)
  g1[2] = (int)(0x2000u << 16);                 // tensor_dim1 = 8192 (lo16)
  g1[3] = (int)(32u << 16);                     // tile_dim0 = 32
  g1[4] = 128;                                  // tile_dim1 = 128 (tile_dim2=0)
  g1[5] = 2048;                                 // tensor_dim0_stride (lo32)
  g1[6] = 0;
  g1[7] = 0;
  v4i z4 = {0, 0, 0, 0};
#if __clang_major__ >= 23
  v8i z8 = {0, 0, 0, 0, 0, 0, 0, 0};
  __builtin_amdgcn_tensor_load_to_lds(g0, g1, z4, z4, z8, 0);
#else
  __builtin_amdgcn_tensor_load_to_lds(g0, g1, z4, z4, 0);
#endif
}
#endif  // HAVE_TDM

// ---------------------------------------------------------------------------
// Kernel 1: QKV GEMM + RoPE + scatter to (B,H,T,hd)
// Block tile 128(M) x 128(N), 8 waves (2x4), each wave 64x32, K-step 32.
// ---------------------------------------------------------------------------
__global__ __launch_bounds__(256) void qkv_rope_kernel(
    const float* __restrict__ x, const float* __restrict__ sinp,
    const float* __restrict__ cosp, const float* __restrict__ Wqkv,
    __bf16* __restrict__ qb, __bf16* __restrict__ kb, __bf16* __restrict__ vb)
{
  __shared__ __attribute__((aligned(64))) __bf16 As[128 * 48];  // [m][k], pitch 48
  __shared__ __attribute__((aligned(64))) __bf16 Bt[128 * 48];  // [n][k], pitch 48

  const int tid  = threadIdx.x;
  const int wave = tid >> 5;
  const int lane = tid & 31;
  const int hl   = lane >> 4;   // half-wave
  const int ln   = lane & 15;
  const int wm   = wave >> 2;   // 0..1
  const int wn   = wave & 3;    // 0..3
  const int n0   = blockIdx.x * 128;
  const int m0   = blockIdx.y * 128;

  v8f acc[4][2];
  #pragma unroll
  for (int i = 0; i < 4; i++) {
    v8f z = {};
    acc[i][0] = z; acc[i][1] = z;
  }

  const int sa_m = tid >> 1;          // 0..127
  const int sa_k = (tid & 1) * 16;    // 0 or 16
  const int sb_n = (tid & 31) * 4;    // 0..124
  const int sb_k = tid >> 5;          // 0..7

  for (int kt = 0; kt < CH; kt += 32) {
    __syncthreads();
    // ---- stage A tile (x, f32 -> bf16), row-major [m][k] ----
    {
      const float* src = x + (size_t)(m0 + sa_m) * CH + kt + sa_k;
      v8bf t0, t1;
      #pragma unroll
      for (int i = 0; i < 8; i++) t0[i] = f2bf(src[i]);
      #pragma unroll
      for (int i = 0; i < 8; i++) t1[i] = f2bf(src[8 + i]);
      *(v8bf*)(&As[sa_m * 48 + sa_k])     = t0;
      *(v8bf*)(&As[sa_m * 48 + sa_k + 8]) = t1;
    }
    // ---- stage B tile (W_qkv, f32 -> bf16), transposed [n][k] ----
    #pragma unroll
    for (int r = 0; r < 4; r++) {
      const int k = sb_k + r * 8;
      const float4 w4 =
          *(const float4*)(Wqkv + (size_t)(kt + k) * NQKV + n0 + sb_n);
      Bt[(sb_n + 0) * 48 + k] = f2bf(w4.x);
      Bt[(sb_n + 1) * 48 + k] = f2bf(w4.y);
      Bt[(sb_n + 2) * 48 + k] = f2bf(w4.z);
      Bt[(sb_n + 3) * 48 + k] = f2bf(w4.w);
    }
    if (kt + 32 < CH) {  // prefetch next tiles -> global_prefetch_b8
      __builtin_prefetch(x + (size_t)(m0 + sa_m) * CH + kt + 32 + sa_k, 0, 0);
      __builtin_prefetch(Wqkv + (size_t)(kt + 32 + sb_k) * NQKV + n0 + sb_n, 0, 0);
    }
    __syncthreads();

    // ---- fragments + WMMA ----
    v16bf bf0 = *(const v16bf*)(&Bt[(wn * 32 + ln) * 48 + 16 * hl]);
    v16bf bf1 = *(const v16bf*)(&Bt[(wn * 32 + 16 + ln) * 48 + 16 * hl]);
    #pragma unroll
    for (int ms = 0; ms < 4; ms++) {
      const int m = wm * 64 + ms * 16 + ln;
      v8bf lo = *(const v8bf*)(&As[m * 48 + 16 * hl]);
      v8bf hi = *(const v8bf*)(&As[m * 48 + 16 * hl + 8]);
      v16bf af = cat8(lo, hi);
      acc[ms][0] = wmma_bf16(af, bf0, acc[ms][0]);
      acc[ms][1] = wmma_bf16(af, bf1, acc[ms][1]);
    }
  }

  // ---- epilogue: RoPE (q,k) and scatter to (B,H,T,hd) ----
  #pragma unroll
  for (int ms = 0; ms < 4; ms++) {
    #pragma unroll
    for (int ns = 0; ns < 2; ns++) {
      const int n    = n0 + wn * 32 + ns * 16 + ln;
      const int sect = n >> 11;          // 0=q 1=k 2=v (uniform per wave)
      const int rem  = n & 2047;
      const int h    = rem >> 7;
      const int d    = rem & 127;
      __bf16* dst = (sect == 0) ? qb : (sect == 1 ? kb : vb);
      #pragma unroll
      for (int i = 0; i < 8; i++) {
        const int m = m0 + wm * 64 + ms * 16 + 8 * hl + i;
        const int b = m >> 11;
        const int t = m & 2047;
        float v = acc[ms][ns][i];
        float outv;
        if (sect < 2) {
          // partner column d^1 lives in lane^1 of the C fragment
          float vp = __shfl_xor(v, 1, 32);
          float sf = sinp[t * HD + d];
          float cf = cosp[t * HD + d];
          outv = v * cf + ((d & 1) ? sf * vp : -sf * vp);
        } else {
          outv = v;
        }
        dst[(size_t)((b * NH + h) * SEQ + t) * HD + d] = f2bf(outv);
      }
    }
  }
}

// ---------------------------------------------------------------------------
// Kernel 2: flash attention, causal. Block = (B,H, 128 queries), 8 waves,
// each wave owns 16 query rows; keys streamed in tiles of 32.
// ---------------------------------------------------------------------------
__global__ __launch_bounds__(256) void attn_kernel(
    const __bf16* __restrict__ qb, const __bf16* __restrict__ kb,
    const __bf16* __restrict__ vb, __bf16* __restrict__ yb)
{
  __shared__ __attribute__((aligned(64))) __bf16 Vt[HD * 48];     // [d][key]
  __shared__ __attribute__((aligned(64))) __bf16 Ps[8 * 16 * 32]; // per-wave P

  const int tid  = threadIdx.x;
  const int wave = tid >> 5;
  const int lane = tid & 31;
  const int hl   = lane >> 4;
  const int ln   = lane & 15;
  const int b    = blockIdx.z;
  const int h    = blockIdx.y;
  const int q0   = blockIdx.x * 128;
  const int qw0  = q0 + wave * 16;
  const size_t headoff = (size_t)(b * NH + h) * SEQ * HD;

  // preload Q fragments (A-layout: lane holds query row ln, 32-wide K chunks)
  v16bf qf[4];
  {
    const __bf16* qrow = qb + headoff + (size_t)(qw0 + ln) * HD;
    #pragma unroll
    for (int dc = 0; dc < 4; dc++) {
      v8bf lo = *(const v8bf*)(qrow + dc * 32 + 16 * hl);
      v8bf hi = *(const v8bf*)(qrow + dc * 32 + 16 * hl + 8);
      qf[dc] = cat8(lo, hi);
    }
  }

  v8f o[8];
  #pragma unroll
  for (int j = 0; j < 8; j++) { v8f z = {}; o[j] = z; }
  float mrow[8], lrow[8];
  #pragma unroll
  for (int i = 0; i < 8; i++) { mrow[i] = -3.0e38f; lrow[i] = 0.0f; }

  const int nkt  = q0 / 32 + 4;        // key tiles needed for this block
  const int sv_d = (tid & 15) * 8;
  const int sv_k = tid >> 4;           // 0..15
  const float scale = 0.088388347648318447f;   // 1/sqrt(128)

  for (int kt = 0; kt < nkt; kt++) {
    const int k0 = kt * 32;
    __syncthreads();
    // ---- stage V tile transposed into LDS: Vt[d][key] ----
    #pragma unroll
    for (int r = 0; r < 2; r++) {
      const int key = sv_k + r * 16;
      v8bf vv = *(const v8bf*)(vb + headoff + (size_t)(k0 + key) * HD + sv_d);
      #pragma unroll
      for (int j = 0; j < 8; j++) Vt[(sv_d + j) * 48 + key] = vv[j];
    }
    __syncthreads();

    if (k0 <= qw0 + 15) {   // wave participates (causal)
      // ---- S = Q @ K^T (two 16x16 key subtiles, K-dim = hd via 4 WMMAs) ----
      v8f s0 = {}, s1 = {};
      #pragma unroll
      for (int dc = 0; dc < 4; dc++) {
        v16bf kf0 = *(const v16bf*)(kb + headoff + (size_t)(k0 + ln) * HD +
                                    dc * 32 + 16 * hl);
        v16bf kf1 = *(const v16bf*)(kb + headoff + (size_t)(k0 + 16 + ln) * HD +
                                    dc * 32 + 16 * hl);
        s0 = wmma_bf16(qf[dc], kf0, s0);
        s1 = wmma_bf16(qf[dc], kf1, s1);
      }

      // ---- online softmax (row = 8*hl + i, cols spread over 16 lanes) ----
      float p0[8], p1[8];
      #pragma unroll
      for (int i = 0; i < 8; i++) {
        const int qrow = qw0 + 8 * hl + i;
        float a0 = s0[i] * scale; if (k0 + ln      > qrow) a0 = -3.0e38f;
        float a1 = s1[i] * scale; if (k0 + 16 + ln > qrow) a1 = -3.0e38f;
        float rm = fmaxf(a0, a1);
        rm = fmaxf(rm, __shfl_xor(rm, 1, 32));
        rm = fmaxf(rm, __shfl_xor(rm, 2, 32));
        rm = fmaxf(rm, __shfl_xor(rm, 4, 32));
        rm = fmaxf(rm, __shfl_xor(rm, 8, 32));
        const float mnew  = fmaxf(mrow[i], rm);
        const float alpha = __expf(mrow[i] - mnew);
        const float e0 = __expf(a0 - mnew);
        const float e1 = __expf(a1 - mnew);
        float rs = e0 + e1;
        rs += __shfl_xor(rs, 1, 32);
        rs += __shfl_xor(rs, 2, 32);
        rs += __shfl_xor(rs, 4, 32);
        rs += __shfl_xor(rs, 8, 32);
        lrow[i] = lrow[i] * alpha + rs;
        mrow[i] = mnew;
        p0[i] = e0; p1[i] = e1;
        #pragma unroll
        for (int j = 0; j < 8; j++) o[j][i] *= alpha;
      }

      // ---- C-layout P -> LDS -> A-layout fragment ----
      __bf16* pw = &Ps[wave * 512];
      #pragma unroll
      for (int i = 0; i < 8; i++) {
        pw[(8 * hl + i) * 32 + ln]      = f2bf(p0[i]);
        pw[(8 * hl + i) * 32 + 16 + ln] = f2bf(p1[i]);
      }
      asm volatile("s_wait_dscnt 0" ::: "memory");  // wave-local LDS RAW fence
      v8bf plo = *(const v8bf*)(pw + ln * 32 + 16 * hl);
      v8bf phi = *(const v8bf*)(pw + ln * 32 + 16 * hl + 8);
      v16bf pf = cat8(plo, phi);

      // ---- O += P @ V (8 d-subtiles) ----
      #pragma unroll
      for (int jd = 0; jd < 8; jd++) {
        v16bf vf = *(const v16bf*)(&Vt[(jd * 16 + ln) * 48 + 16 * hl]);
        o[jd] = wmma_bf16(pf, vf, o[jd]);
      }
    }
  }

  // ---- normalize and write y (bf16, (B*T, C) layout, c = h*hd + d) ----
  #pragma unroll
  for (int i = 0; i < 8; i++) {
    const float rl = 1.0f / lrow[i];
    const int   t  = qw0 + 8 * hl + i;
    __bf16* dst = yb + (size_t)(b * SEQ + t) * CH + h * HD + ln;
    #pragma unroll
    for (int jd = 0; jd < 8; jd++) dst[jd * 16] = f2bf(o[jd][i] * rl);
  }
}

// ---------------------------------------------------------------------------
// Kernel 3: out = y @ W_proj (f32 output). Same tiling as kernel 1, but the
// A tile (already bf16) is staged LDS-side by the Tensor Data Mover.
// ---------------------------------------------------------------------------
__global__ __launch_bounds__(256) void proj_kernel(
    const __bf16* __restrict__ yb, const float* __restrict__ Wp,
    float* __restrict__ out)
{
  __shared__ __attribute__((aligned(64))) __bf16 As[128 * 48];
  __shared__ __attribute__((aligned(64))) __bf16 Bt[128 * 48];

  const int tid  = threadIdx.x;
  const int wave = tid >> 5;
  const int lane = tid & 31;
  const int hl   = lane >> 4;
  const int ln   = lane & 15;
  const int wm   = wave >> 2;
  const int wn   = wave & 3;
  const int n0   = blockIdx.x * 128;
  const int m0   = blockIdx.y * 128;

  v8f acc[4][2];
  #pragma unroll
  for (int i = 0; i < 4; i++) { v8f z = {}; acc[i][0] = z; acc[i][1] = z; }

  const int sb_n = (tid & 31) * 4;
  const int sb_k = tid >> 5;
#if !HAVE_TDM
  const int sa_m = tid >> 1;
  const int sa_k = (tid & 1) * 16;
#endif

  for (int kt = 0; kt < CH; kt += 32) {
    __syncthreads();
#if HAVE_TDM
    // ---- stage A via TDM (wave 0 issues; DMA writes pitch-48 LDS image) ----
    if (wave == 0) {
      tdm_load_tile_128x32(yb + (size_t)m0 * CH + kt,
                           (unsigned)(uintptr_t)(&As[0]));
    }
#else
    {  // fallback: stage A by hand (already bf16)
      const __bf16* src = yb + (size_t)(m0 + sa_m) * CH + kt + sa_k;
      v8bf t0 = *(const v8bf*)(src);
      v8bf t1 = *(const v8bf*)(src + 8);
      *(v8bf*)(&As[sa_m * 48 + sa_k])     = t0;
      *(v8bf*)(&As[sa_m * 48 + sa_k + 8]) = t1;
    }
#endif
    #pragma unroll
    for (int r = 0; r < 4; r++) {  // stage B transposed, f32 -> bf16
      const int k = sb_k + r * 8;
      const float4 w4 = *(const float4*)(Wp + (size_t)(kt + k) * CH + n0 + sb_n);
      Bt[(sb_n + 0) * 48 + k] = f2bf(w4.x);
      Bt[(sb_n + 1) * 48 + k] = f2bf(w4.y);
      Bt[(sb_n + 2) * 48 + k] = f2bf(w4.z);
      Bt[(sb_n + 3) * 48 + k] = f2bf(w4.w);
    }
    if (kt + 32 < CH) {
      __builtin_prefetch(Wp + (size_t)(kt + 32 + sb_k) * CH + n0 + sb_n, 0, 0);
    }
#if HAVE_TDM
    if (wave == 0) __builtin_amdgcn_s_wait_tensorcnt(0);
#endif
    __syncthreads();

    v16bf bf0 = *(const v16bf*)(&Bt[(wn * 32 + ln) * 48 + 16 * hl]);
    v16bf bf1 = *(const v16bf*)(&Bt[(wn * 32 + 16 + ln) * 48 + 16 * hl]);
    #pragma unroll
    for (int ms = 0; ms < 4; ms++) {
      const int m = wm * 64 + ms * 16 + ln;
      v8bf lo = *(const v8bf*)(&As[m * 48 + 16 * hl]);
      v8bf hi = *(const v8bf*)(&As[m * 48 + 16 * hl + 8]);
      v16bf af = cat8(lo, hi);
      acc[ms][0] = wmma_bf16(af, bf0, acc[ms][0]);
      acc[ms][1] = wmma_bf16(af, bf1, acc[ms][1]);
    }
  }

  #pragma unroll
  for (int ms = 0; ms < 4; ms++) {
    #pragma unroll
    for (int ns = 0; ns < 2; ns++) {
      const int n = n0 + wn * 32 + ns * 16 + ln;
      #pragma unroll
      for (int i = 0; i < 8; i++) {
        const int m = m0 + wm * 64 + ms * 16 + 8 * hl + i;
        out[(size_t)m * CH + n] = acc[ms][ns][i];
      }
    }
  }
}

// ---------------------------------------------------------------------------
extern "C" void kernel_launch(void* const* d_in, const int* in_sizes, int n_in,
                              void* d_out, int out_size, void* d_ws,
                              size_t ws_size, hipStream_t stream) {
  const float* x     = (const float*)d_in[0];
  const float* sinp  = (const float*)d_in[1];
  const float* cosp  = (const float*)d_in[2];
  const float* Wqkv  = (const float*)d_in[3];
  const float* Wproj = (const float*)d_in[4];
  float* out = (float*)d_out;

  const size_t per = (size_t)BATCH * NH * SEQ * HD;  // 16,777,216 elems
  __bf16* qb = (__bf16*)d_ws;
  __bf16* kb = qb + per;
  __bf16* vb = kb + per;
  __bf16* yb = vb + per;  // total 128 MB of bf16 workspace

  qkv_rope_kernel<<<dim3(NQKV / 128, (BATCH * SEQ) / 128), 256, 0, stream>>>(
      x, sinp, cosp, Wqkv, qb, kb, vb);
  attn_kernel<<<dim3(SEQ / 128, NH, BATCH), 256, 0, stream>>>(qb, kb, vb, yb);
  proj_kernel<<<dim3(CH / 128, (BATCH * SEQ) / 128), 256, 0, stream>>>(
      yb, Wproj, out);
}
